// HouseholderTower_65163243815687
// MI455X (gfx1250) — matrix operand order repaired and checked by reading, not compiled, not run on hardware
//
#include <hip/hip_runtime.h>
#include <hip/hip_bf16.h>

// CDNA5 WMMA vector types (wave32): 16 bf16 = 8 VGPRs, 8 f32 = 8 VGPRs.
typedef __attribute__((ext_vector_type(16))) __bf16 v16bf;
typedef __attribute__((ext_vector_type(8)))  float  v8f;

// 32-byte blob <-> v16bf fragment (element e = bytes [2e, 2e+1]).
union Frag32B {
    uint4  q[2];
    v16bf  v;
};

// ---------------------------------------------------------------------------
// Kernel 1: fold the 16 Householder reflections into one 256x256 matrix M,
// then emit it transposed ([n][k], k-contiguous) as split-bf16 (hi + lo).
// One workgroup, thread t owns row k=t of M (kept in workspace global mem).
// ---------------------------------------------------------------------------
__global__ void hh_build_M(const float* __restrict__ V,   // [16][256]
                           float* __restrict__ Mf,        // ws: [256][256] fp32 scratch
                           unsigned short* __restrict__ MhiT,  // ws: [256][256] bf16 bits, transposed
                           unsigned short* __restrict__ MloT)  // ws: [256][256] bf16 bits, transposed
{
    const int t = threadIdx.x;            // 0..255 : row index k of M
    float* row = Mf + t * 256;

    // M = I
    for (int j = 0; j < 256; ++j) row[j] = (j == t) ? 1.0f : 0.0f;

    __shared__ float v[256];
    __shared__ float vvinv;

    for (int r = 0; r < 16; ++r) {
        __syncthreads();                  // protect v[] from previous round readers
        v[t] = V[r * 256 + t];
        __syncthreads();
        if (t == 0) {
            float s = 0.0f;
            for (int j = 0; j < 256; ++j) s += v[j] * v[j];
            vvinv = 1.0f / s;
        }
        __syncthreads();
        // row_k <- row_k - ((row_k . v) / (v.v)) * v
        float s = 0.0f;
        for (int j = 0; j < 256; ++j) s += row[j] * v[j];
        const float c = s * vvinv;
        for (int j = 0; j < 256; ++j) row[j] -= c * v[j];
    }

    // Emit transposed split-bf16: MT[n][k] = M[k][n].  hi = RNE(x), lo = RNE(x - hi).
    for (int j = 0; j < 256; ++j) {
        const float x  = row[j];
        const __bf16 h = (__bf16)x;
        const __bf16 l = (__bf16)(x - (float)h);
        MhiT[j * 256 + t] = __builtin_bit_cast(unsigned short, h);
        MloT[j * 256 + t] = __builtin_bit_cast(unsigned short, l);
    }
}

// ---------------------------------------------------------------------------
// Kernel 2: out[N,256] = X[N,256] @ M[256,256] via split-bf16 WMMA.
// Block = 256 threads (8 waves) handles 32 rows x 256 cols.
// Wave w owns n-columns [32w, 32w+32) and both 16-row tiles:
//   4 accumulator tiles (2 row-tiles x 2 col-tiles), full K=256.
// Per K-step: 2 A hi/lo fragment pairs (LDS) + 2 B hi/lo pairs (global/L2),
// each B fragment feeds both row tiles -> 12 WMMA per K-step, 96 per wave.
// X is read from HBM exactly once; B/L2 traffic = N/32 * 256KB = 1.6 GB.
// ---------------------------------------------------------------------------
#define ROWS_PER_BLOCK 32
#define LDS_PITCH 264   // 256 + 8 bf16 pad: row stride 528 B -> conflict-free b128 reads

__global__ void hh_gemm(const float* __restrict__ X,
                        const unsigned short* __restrict__ MhiT,
                        const unsigned short* __restrict__ MloT,
                        float* __restrict__ out,
                        int nrows)
{
    __shared__ unsigned short Ahi[ROWS_PER_BLOCK * LDS_PITCH];
    __shared__ unsigned short Alo[ROWS_PER_BLOCK * LDS_PITCH];

    const int tid  = threadIdx.x;
    const int row0 = blockIdx.x * ROWS_PER_BLOCK;
    if (row0 + ROWS_PER_BLOCK > nrows) return;

    // ---- cooperative load: 32x256 fp32 slab -> split-bf16 in LDS ----
    const float4* src = (const float4*)(X + (size_t)row0 * 256);
    #pragma unroll
    for (int it = 0; it < 8; ++it) {
        const int idx = it * 256 + tid;        // float4 index within slab (0..2047)
        const float4 f = src[idx];
        const int r = (idx * 4) >> 8;          // row 0..31
        const int c = (idx * 4) & 255;         // col (multiple of 4)
        const float xs[4] = {f.x, f.y, f.z, f.w};
        #pragma unroll
        for (int u = 0; u < 4; ++u) {
            const float  x = xs[u];
            const __bf16 h = (__bf16)x;
            const __bf16 l = (__bf16)(x - (float)h);
            Ahi[r * LDS_PITCH + c + u] = __builtin_bit_cast(unsigned short, h);
            Alo[r * LDS_PITCH + c + u] = __builtin_bit_cast(unsigned short, l);
        }
    }
    __syncthreads();

    const int wave = tid >> 5;
    const int lane = tid & 31;
    const int half = lane >> 4;   // 0: lanes 0-15, 1: lanes 16-31
    const int lm   = lane & 15;
    const int n0   = wave * 32;

    v8f acc[2][2] = {{v8f{}, v8f{}}, {v8f{}, v8f{}}};

    #pragma unroll
    for (int k0 = 0; k0 < 256; k0 += 32) {
        // A fragments (16-bit A 16x32 layout) for both 16-row tiles:
        // lane holds row m = rt*16 + lm;
        // VGPR 0-3 = K = half*8 + 0..7, VGPR 4-7 = K = 16 + half*8 + 0..7.
        Frag32B ahi[2], alo[2];
        #pragma unroll
        for (int rt = 0; rt < 2; ++rt) {
            const int offA0 = (rt * 16 + lm) * LDS_PITCH + k0 + half * 8;
            const int offA1 = offA0 + 16;
            ahi[rt].q[0] = *(const uint4*)(Ahi + offA0);
            ahi[rt].q[1] = *(const uint4*)(Ahi + offA1);
            alo[rt].q[0] = *(const uint4*)(Alo + offA0);
            alo[rt].q[1] = *(const uint4*)(Alo + offA1);
        }

        #pragma unroll
        for (int ct = 0; ct < 2; ++ct) {
            // B fragment (16-bit B 32x16 layout): lane holds col n = n0+ct*16+lm;
            // VGPR 0-3 -> K = k0 + half*8 + 0..7, VGPR 4-7 -> K = k0 + 16 + half*8 + 0..7
            // (two 8-bf16 runs of transposed-M row n), matching A.
            const int n = n0 + ct * 16 + lm;
            const uint4* pbh = (const uint4*)(MhiT + n * 256 + k0 + half * 8);
            const uint4* pbl = (const uint4*)(MloT + n * 256 + k0 + half * 8);
            Frag32B bhi, blo;
            bhi.q[0] = pbh[0]; bhi.q[1] = pbh[2];
            blo.q[0] = pbl[0]; blo.q[1] = pbl[2];

            // fp32-split product: Ahi*Bhi + Ahi*Blo + Alo*Bhi (f32 accumulate);
            // B fragment reused across both row tiles.
            #pragma unroll
            for (int rt = 0; rt < 2; ++rt) {
                acc[rt][ct] = __builtin_amdgcn_wmma_f32_16x16x32_bf16(
                    false, ahi[rt].v, false, bhi.v, (short)0, acc[rt][ct], false, false);
                acc[rt][ct] = __builtin_amdgcn_wmma_f32_16x16x32_bf16(
                    false, ahi[rt].v, false, blo.v, (short)0, acc[rt][ct], false, false);
                acc[rt][ct] = __builtin_amdgcn_wmma_f32_16x16x32_bf16(
                    false, alo[rt].v, false, bhi.v, (short)0, acc[rt][ct], false, false);
            }
        }
    }

    // D layout (32-bit C/D 16x16): lane -> col n = lm, VGPR r -> row half*8 + r.
    #pragma unroll
    for (int rt = 0; rt < 2; ++rt) {
        #pragma unroll
        for (int ct = 0; ct < 2; ++ct) {
            #pragma unroll
            for (int r = 0; r < 8; ++r) {
                out[(size_t)(row0 + rt * 16 + half * 8 + r) * 256
                    + (n0 + ct * 16 + lm)] = acc[rt][ct][r];
            }
        }
    }
}

// ---------------------------------------------------------------------------
// Launcher. Inputs: d_in[0]=node_feat f32[N,256], d_in[1]=adj (unused),
// d_in[2]=V f32[16,256].  Output: f32[N,256].
// Workspace: MhiT (128 KB) | MloT (128 KB) | Mf (256 KB)  = 512 KB.
// ---------------------------------------------------------------------------
extern "C" void kernel_launch(void* const* d_in, const int* in_sizes, int n_in,
                              void* d_out, int out_size, void* d_ws, size_t ws_size,
                              hipStream_t stream) {
    const float* node_feat = (const float*)d_in[0];
    const float* V         = (const float*)d_in[2];
    float*       out       = (float*)d_out;

    unsigned short* MhiT = (unsigned short*)d_ws;
    unsigned short* MloT = MhiT + 256 * 256;
    float*          Mf   = (float*)(MloT + 256 * 256);

    const int nrows = in_sizes[0] / 256;                          // N = 200000
    const int grid  = (nrows + ROWS_PER_BLOCK - 1) / ROWS_PER_BLOCK;  // 6250

    hh_build_M<<<1, 256, 0, stream>>>(V, Mf, MhiT, MloT);
    hh_gemm<<<grid, 256, 0, stream>>>(node_feat, MhiT, MloT, out, nrows);
}
